// StreamingSSMController_83434034692785
// MI455X (gfx1250) — matrix operand order repaired and checked by reading, not compiled, not run on hardware
//
#include <hip/hip_runtime.h>
#include <hip/hip_bf16.h>
#include <math.h>

// ---- problem constants ----
#define BB 32
#define TT 4096
#define DD 256
#define LL 2
#define KK 4
#define MROWS 16        // batch rows per block (one WMMA M-tile)
#define NTHREADS 512    // 16 wave32 waves
#define KT 8            // K=256 / 32 per bf16 WMMA

typedef __attribute__((ext_vector_type(16))) __bf16 v16bf;
typedef __attribute__((ext_vector_type(8)))  __bf16 v8bf;
typedef __attribute__((ext_vector_type(8)))  float  v8f;

// bf16 fragment-swizzled weight regions inside d_ws (element offsets)
#define ENC_OFF   0
#define IP_OFF(l) (65536 + (l)*131072)
#define OP_OFF(l) (65536 + 2*131072 + (l)*65536)
#define FRAG_TOTAL (65536 + 2*131072 + 2*65536)   // 458752 bf16 elems

// ---------------------------------------------------------------------------
// One-time (per launch) weight convert + swizzle into WMMA B-fragment order.
// B-matrix (K x N tile = 32 x 16) wave32 layout used here (contiguous k, not the ISA table order):
//   n = lane & 15 ; K = (lane>>4)*16 + h  (h = 0..15 contiguous halves)
// Stored so each lane reads 16 contiguous bf16 (32 bytes).
// ---------------------------------------------------------------------------
__global__ void wconv_kernel(const float* __restrict__ enc_w,
                             const float* __restrict__ ip_w,
                             const float* __restrict__ op_w,
                             __bf16* __restrict__ frag) {
  int idx = blockIdx.x * blockDim.x + threadIdx.x;
  if (idx >= FRAG_TOTAL) return;
  const float* W;
  int base;
  if (idx < 65536)               { W = enc_w;             base = idx; }
  else if (idx < 196608)         { W = ip_w;              base = idx - 65536; }
  else if (idx < 327680)         { W = ip_w + 512 * 256;  base = idx - 196608; }
  else if (idx < 393216)         { W = op_w;              base = idx - 327680; }
  else                           { W = op_w + 256 * 256;  base = idx - 393216; }
  int tile = base >> 9;          // (nt*KT + kt)
  int rem  = base & 511;
  int lane = rem >> 4;
  int h    = rem & 15;
  int nt   = tile / KT;
  int kt   = tile % KT;
  int n = nt * 16 + (lane & 15);
  int k = kt * 32 + ((lane >> 4) * 16) + h;
  frag[idx] = (__bf16)W[n * 256 + k];   // all weights row-major (N,K), K=256
}

__device__ __forceinline__ float silu_f(float v) { return v / (1.0f + expf(-v)); }

// read one A fragment (16x32 bf16) from swizzled LDS: 2x ds_load_b128, no cvt
__device__ __forceinline__ v16bf load_a(const __bf16* abuf, int kt, int lane) {
  const __bf16* p = abuf + ((kt * 32 + lane) << 4);
  v8bf lo = *(const v8bf*)(p);
  v8bf hi = *(const v8bf*)(p + 8);
  v16bf a;
#pragma unroll
  for (int h = 0; h < 8; ++h) { a[h] = lo[h]; a[h + 8] = hi[h]; }
  return a;
}

// pack 8 f32 (one row, 8 consecutive channels) into swizzled A LDS, 16B store
__device__ __forceinline__ void pack_a(__bf16* abuf, int pofs, const float* v) {
  v8bf p;
#pragma unroll
  for (int j = 0; j < 8; ++j) p[j] = (__bf16)v[j];
  *(v8bf*)(abuf + pofs) = p;
}

// ---------------------------------------------------------------------------
// Persistent scan kernel: 2 blocks x 512 threads (16 waves), 16 batch rows
// per block. Recurrent state (conv taps, EMA h-state) lives in registers.
// ---------------------------------------------------------------------------
__global__ void __launch_bounds__(NTHREADS)
ssm_kernel(const float* __restrict__ x,
           const float* __restrict__ enc_b,
           const float* __restrict__ enc_ln_g, const float* __restrict__ enc_ln_b,
           const float* __restrict__ in_proj_b,
           const float* __restrict__ conv_w,  const float* __restrict__ conv_b,
           const float* __restrict__ out_proj_b,
           const float* __restrict__ lnf_g,   const float* __restrict__ lnf_b,
           const __bf16* __restrict__ frag,
           float* __restrict__ out) {
  extern __shared__ char smem_raw[];
  __bf16* abuf = (__bf16*)smem_raw;                   // 4096 bf16 = 8 KB (A frags)
  float*  pre  = (float*)(smem_raw + 8192);           // 16x256 f32 (GEMM1 out)
  float*  hbuf = (float*)(smem_raw + 8192 + 16384);   // 16x256 f32 (h)
  float*  xz   = (float*)(smem_raw + 8192 + 32768);   // 16x512 f32 (in_proj out)
  // total 72 KB dynamic LDS (WGP pool is 320 KB)

  const int tid  = threadIdx.x;
  const int lane = tid & 31;
  const int w    = tid >> 5;              // wave id == batch row for pack phases
  const int bofs = blockIdx.x * MROWS;
  const int d0   = lane * 8;              // this thread's 8 channels (row w)
  // swizzled-A destination for this thread's 8 consecutive K values of row w:
  //   kt = lane>>2, lane'hi = (lane>>1)&1, h0 = (lane&1)*8
  const int pofs = (((lane >> 2) * 32) + (((lane >> 1) & 1) * 16) + w) * 16
                 + (lane & 1) * 8;

  // ---- hoist per-thread constants (same every timestep) ----
  float elng[8], elnb[8], flng[8], flnb[8], dec[8];
  float cwr[LL][8][KK], cbr[LL][8];
#pragma unroll
  for (int j = 0; j < 8; ++j) {
    int d = d0 + j;
    elng[j] = enc_ln_g[d]; elnb[j] = enc_ln_b[d];
    flng[j] = lnf_g[d];    flnb[j] = lnf_b[d];
    float tl = logf(10.0f) + (logf(2000.0f) - logf(10.0f)) * ((float)d / 255.0f);
    dec[j]  = expf(-1.0f / expf(tl));
#pragma unroll
    for (int l = 0; l < LL; ++l) {
      cbr[l][j] = conv_b[l * DD + d];
#pragma unroll
      for (int k = 0; k < KK; ++k) cwr[l][j][k] = conv_w[(l * DD + d) * KK + k];
    }
  }
  // per-lane GEMM epilogue constants (C/D layout: n = lane&15, m = (lane>>4)*8 + r)
  const int ncol  = lane & 15;
  const int mrow0 = (lane >> 4) * 8;
  const float encb = enc_b[w * 16 + ncol];
  float ipb0[LL], ipb1[LL], opb[LL];
#pragma unroll
  for (int l = 0; l < LL; ++l) {
    ipb0[l] = in_proj_b[l * 2 * DD + w * 16 + ncol];
    ipb1[l] = in_proj_b[l * 2 * DD + DD + w * 16 + ncol];
    opb[l]  = out_proj_b[l * DD + w * 16 + ncol];
  }

  // ---- recurrent state in registers ----
  float q[LL][8][KK - 1];   // last 3 conv inputs (oldest first)
  float hs[LL][8];          // EMA state
#pragma unroll
  for (int l = 0; l < LL; ++l)
#pragma unroll
    for (int j = 0; j < 8; ++j) {
      hs[l][j] = 0.f; q[l][j][0] = 0.f; q[l][j][1] = 0.f; q[l][j][2] = 0.f;
    }

  // Opaque ZERO OFFSET, redefined every timestep: blocks LICM from hoisting the
  // loop-invariant B-fragment loads (no spills), while keeping `frag`'s global
  // address-space provenance intact so the loads stay global_load (LOADcnt
  // only), not flat_load (LOADcnt+DScnt false serialization with LDS A loads).
  size_t fofs = 0;

  for (int t = 0; t < TT; ++t) {
    asm volatile("" : "+s"(fofs));
    const __bf16* fp = frag + fofs;

    // ---- stage x_t: global f32 -> swizzled bf16 A frags, prefetch t+1 ----
    {
      const float* src = x + ((size_t)(bofs + w) * TT + t) * DD + d0;
      float v[8];
      float4 v0 = *(const float4*)(src);
      float4 v1 = *(const float4*)(src + 4);
      v[0]=v0.x; v[1]=v0.y; v[2]=v0.z; v[3]=v0.w;
      v[4]=v1.x; v[5]=v1.y; v[6]=v1.z; v[7]=v1.w;
      pack_a(abuf, pofs, v);
      if (t + 1 < TT) __builtin_prefetch(src + DD, 0, 1);
    }
    __syncthreads();

    // ---- GEMM1: pre = x_t @ enc_w^T + enc_b  (M=16,N=256,K=256) ----
    {
      v8f acc = {};
#pragma unroll
      for (int kt = 0; kt < KT; ++kt) {
        v16bf a = load_a(abuf, kt, lane);
        v16bf b = *(const v16bf*)(fp + ENC_OFF + ((w * KT + kt) << 9) + (lane << 4));
        acc = __builtin_amdgcn_wmma_f32_16x16x32_bf16(false, a, false, b,
                                                      (short)0, acc, false, false);
      }
#pragma unroll
      for (int r = 0; r < 8; ++r)
        pre[(mrow0 + r) * DD + w * 16 + ncol] = acc[r] + encb;
    }
    __syncthreads();

    // ---- LN + ReLU -> hbuf (f32) + swizzled A frags ----
    {
      float v[8]; float s = 0.f;
#pragma unroll
      for (int j = 0; j < 8; ++j) { v[j] = pre[w * DD + d0 + j]; s += v[j]; }
#pragma unroll
      for (int off = 16; off > 0; off >>= 1) s += __shfl_xor(s, off, 32);
      float mean = s * (1.0f / DD);
      float sq = 0.f;
#pragma unroll
      for (int j = 0; j < 8; ++j) { float c = v[j] - mean; sq += c * c; }
#pragma unroll
      for (int off = 16; off > 0; off >>= 1) sq += __shfl_xor(sq, off, 32);
      float rstd = rsqrtf(sq * (1.0f / DD) + 1e-5f);
#pragma unroll
      for (int j = 0; j < 8; ++j) {
        float hv = (v[j] - mean) * rstd * elng[j] + elnb[j];
        v[j] = hv > 0.f ? hv : 0.f;
        hbuf[w * DD + d0 + j] = v[j];
      }
      pack_a(abuf, pofs, v);
    }
    __syncthreads();

#pragma unroll
    for (int l = 0; l < LL; ++l) {
      // ---- GEMM2: xz = h @ in_proj^T + b  (M=16,N=512,K=256; 2 N-tiles/wave) ----
      {
        v8f acc0 = {}, acc1 = {};
#pragma unroll
        for (int kt = 0; kt < KT; ++kt) {
          v16bf a  = load_a(abuf, kt, lane);
          v16bf b0 = *(const v16bf*)(fp + IP_OFF(l) + ((w * KT + kt) << 9) + (lane << 4));
          v16bf b1 = *(const v16bf*)(fp + IP_OFF(l) + (((w + 16) * KT + kt) << 9) + (lane << 4));
          acc0 = __builtin_amdgcn_wmma_f32_16x16x32_bf16(false, a, false, b0,
                                                         (short)0, acc0, false, false);
          acc1 = __builtin_amdgcn_wmma_f32_16x16x32_bf16(false, a, false, b1,
                                                         (short)0, acc1, false, false);
        }
#pragma unroll
        for (int r = 0; r < 8; ++r) {
          xz[(mrow0 + r) * (2 * DD) + w * 16 + ncol]      = acc0[r] + ipb0[l];
          xz[(mrow0 + r) * (2 * DD) + DD + w * 16 + ncol] = acc1[r] + ipb1[l];
        }
      }
      __syncthreads();

      // ---- depthwise conv (K=4) + EMA state + SiLU gate -> packed g frags ----
      {
        float g[8];
#pragma unroll
        for (int j = 0; j < 8; ++j) {
          float xin = xz[w * (2 * DD) + d0 + j];
          float z   = xz[w * (2 * DD) + DD + d0 + j];
          float c = q[l][j][0] * cwr[l][j][0] + q[l][j][1] * cwr[l][j][1]
                  + q[l][j][2] * cwr[l][j][2] + xin * cwr[l][j][3] + cbr[l][j];
          float co = silu_f(c);
          q[l][j][0] = q[l][j][1]; q[l][j][1] = q[l][j][2]; q[l][j][2] = xin;
          float hi = dec[j] * hs[l][j] + (1.0f - dec[j]) * co;
          hs[l][j] = hi;
          g[j] = hi * silu_f(z);
        }
        pack_a(abuf, pofs, g);
      }
      __syncthreads();

      // ---- GEMM3: h += g @ out_proj^T + b  (M=16,N=256,K=256) ----
      {
        v8f acc = {};
#pragma unroll
        for (int kt = 0; kt < KT; ++kt) {
          v16bf a = load_a(abuf, kt, lane);
          v16bf b = *(const v16bf*)(fp + OP_OFF(l) + ((w * KT + kt) << 9) + (lane << 4));
          acc = __builtin_amdgcn_wmma_f32_16x16x32_bf16(false, a, false, b,
                                                        (short)0, acc, false, false);
        }
#pragma unroll
        for (int r = 0; r < 8; ++r)
          hbuf[(mrow0 + r) * DD + w * 16 + ncol] += acc[r] + opb[l];
      }
      __syncthreads();

      // ---- re-pack updated h into A frags for the next layer's GEMM2 ----
      if (l + 1 < LL) {
        const float* hr = hbuf + w * DD + d0;
        float v[8];
        float4 v0 = *(const float4*)(hr);
        float4 v1 = *(const float4*)(hr + 4);
        v[0]=v0.x; v[1]=v0.y; v[2]=v0.z; v[3]=v0.w;
        v[4]=v1.x; v[5]=v1.y; v[6]=v1.z; v[7]=v1.w;
        pack_a(abuf, pofs, v);
        __syncthreads();
      }
    }

    // ---- final LN -> global output ----
    {
      float v[8]; float s = 0.f;
#pragma unroll
      for (int j = 0; j < 8; ++j) { v[j] = hbuf[w * DD + d0 + j]; s += v[j]; }
#pragma unroll
      for (int off = 16; off > 0; off >>= 1) s += __shfl_xor(s, off, 32);
      float mean = s * (1.0f / DD);
      float sq = 0.f;
#pragma unroll
      for (int j = 0; j < 8; ++j) { float c = v[j] - mean; sq += c * c; }
#pragma unroll
      for (int off = 16; off > 0; off >>= 1) sq += __shfl_xor(sq, off, 32);
      float rstd = rsqrtf(sq * (1.0f / DD) + 1e-5f);
      float* o = out + ((size_t)(bofs + w) * TT + t) * DD + d0;
#pragma unroll
      for (int j = 0; j < 8; ++j)
        o[j] = (v[j] - mean) * rstd * flng[j] + flnb[j];
    }
    __syncthreads();
  }
}

extern "C" void kernel_launch(void* const* d_in, const int* in_sizes, int n_in,
                              void* d_out, int out_size, void* d_ws, size_t ws_size,
                              hipStream_t stream) {
  const float* x        = (const float*)d_in[0];
  const float* enc_w    = (const float*)d_in[1];
  const float* enc_b    = (const float*)d_in[2];
  const float* enc_ln_g = (const float*)d_in[3];
  const float* enc_ln_b = (const float*)d_in[4];
  const float* ip_w     = (const float*)d_in[5];
  const float* ip_b     = (const float*)d_in[6];
  const float* conv_w   = (const float*)d_in[7];
  const float* conv_b   = (const float*)d_in[8];
  const float* op_w     = (const float*)d_in[9];
  const float* op_b     = (const float*)d_in[10];
  const float* lnf_g    = (const float*)d_in[11];
  const float* lnf_b    = (const float*)d_in[12];
  __bf16* frag = (__bf16*)d_ws;          // ~0.9 MB bf16 swizzled weights
  float* out = (float*)d_out;

  wconv_kernel<<<(FRAG_TOTAL + 255) / 256, 256, 0, stream>>>(enc_w, ip_w, op_w, frag);
  ssm_kernel<<<BB / MROWS, NTHREADS, 73728, stream>>>(
      x, enc_b, enc_ln_g, enc_ln_b, ip_b, conv_w, conv_b, op_b, lnf_g, lnf_b,
      frag, out);
}
